// EGNN_encoder_QM9_26396869001241
// MI455X (gfx1250) — matrix operand order repaired and checked by reading, not compiled, not run on hardware
//
#include <hip/hip_runtime.h>
#include <math.h>

// ===========================================================================
// EGNN encoder forward for MI455X (gfx1250, wave32, WMMA 16x16x32 f16->f32)
// - weights pre-packed into B-fragment layout (one 32B load per lane per frag)
// - B fragments prefetched per K-step so loads clause together and the 8-WMMA
//   burst hides the load latency
// - no data-dependent branches inside the WMMA loops
// ===========================================================================

typedef __attribute__((ext_vector_type(16))) _Float16 v16h;
typedef __attribute__((ext_vector_type(8)))  _Float16 v8h;
typedef __attribute__((ext_vector_type(8)))  float    v8f;
typedef __attribute__((ext_vector_type(4)))  float    v4f;

#define NB      16          // batches
#define NN      96          // nodes per batch
#define BNODES  (NB * NN)   // 1536
#define HID     128
#define NEDGE   (NN * NN)   // edges per batch (9216)
#define NORMV   100.0f

__device__ __forceinline__ float silu_f(float x) {
  return x / (1.0f + __expf(-x));
}

// ---------------------------------------------------------------------------
// WMMA fragment helpers (wave32, V_WMMA_F32_16X16X32_F16)
// A (16M x 32K f16): lane&15 = M row; lane>>4 selects K interleave half:
//   VGPR p<8  -> k = 8*half + p ;  p>=8 -> k = 16 + 8*half + (p-8)
//   => two contiguous 8-element runs per lane
// B (32K x 16N f16): pre-packed (see pack_w_kernel): frag = Wp[(ks*8+nt)*32+lane]
// C/D (16x16 f32, 8 VGPRs): lane&15 = N col; row m = v + 8*(lane>>4)
// ---------------------------------------------------------------------------
__device__ __forceinline__ v16h frag_a_from_f32(const float* __restrict__ A,
                                                int ld, int k0, int lane) {
  const int m = lane & 15, hf = lane >> 4;
  const float* r = A + m * ld + k0;
  v4f a0 = *(const v4f*)(r + 8 * hf);
  v4f a1 = *(const v4f*)(r + 8 * hf + 4);
  v4f b0 = *(const v4f*)(r + 16 + 8 * hf);
  v4f b1 = *(const v4f*)(r + 16 + 8 * hf + 4);
  v16h a;
#pragma unroll
  for (int p = 0; p < 4; ++p) {
    a[p]      = (_Float16)a0[p];
    a[p + 4]  = (_Float16)a1[p];
    a[p + 8]  = (_Float16)b0[p];
    a[p + 12] = (_Float16)b1[p];
  }
  return a;
}

__device__ __forceinline__ v16h frag_a_from_h16(const _Float16* __restrict__ T,
                                                int ld, int k0, int lane) {
  const int m = lane & 15, hf = lane >> 4;
  const _Float16* r = T + m * ld + k0;
  v8h lo = *(const v8h*)(r + 8 * hf);        // 16B aligned -> ds_load_b128
  v8h hi = *(const v8h*)(r + 16 + 8 * hf);
  v16h a;
#pragma unroll
  for (int p = 0; p < 8; ++p) {
    a[p]     = lo[p];
    a[p + 8] = hi[p];
  }
  return a;
}

// ---------------------------------------------------------------------------
// Pack a (stack of) 128x128 f32 weight matrices into B-fragment layout:
//   out[mat][(ks*8+nt)*32+lane][p] =
//       (f16) W[mat][ks*32 + (p<8 ? 8*(lane>>4)+p : 16+8*(lane>>4)+(p-8))][nt*16 + (lane&15)]
// ---------------------------------------------------------------------------
__global__ void pack_w_kernel(const float* __restrict__ W, size_t in_stride,
                              v16h* __restrict__ out, int nmat) {
  int idx = blockIdx.x * blockDim.x + threadIdx.x;
  int total = nmat * 32 * 32;
  if (idx >= total) return;
  int mat = idx >> 10;
  int rem = idx & 1023;
  int frag = rem >> 5;
  int lane = rem & 31;
  int ks = frag >> 3, nt = frag & 7;
  int n = lane & 15, hf = lane >> 4;
  const float* Wm = W + (size_t)mat * in_stride;
  v16h v;
#pragma unroll
  for (int p = 0; p < 16; ++p) {
    int k = ks * 32 + (p < 8 ? 8 * hf + p : 16 + 8 * hf + (p - 8));
    v[p] = (_Float16)Wm[(size_t)k * HID + nt * 16 + n];
  }
  out[(size_t)mat * 1024 + rem] = v;
}

// ---------------------------------------------------------------------------
// Embedding + x init
// ---------------------------------------------------------------------------
__global__ void embed_init_kernel(const float* __restrict__ xh,
                                  const float* __restrict__ nmask,
                                  const float* __restrict__ ctx,
                                  const float* __restrict__ ew,
                                  const float* __restrict__ eb,
                                  float* __restrict__ h,
                                  float* __restrict__ x0,
                                  float* __restrict__ xb0) {
  int idx = blockIdx.x * blockDim.x + threadIdx.x;
  if (idx >= BNODES * HID) return;
  int node = idx >> 7;
  int c = idx & (HID - 1);
  float m = nmask[node];
  float in7[7];
#pragma unroll
  for (int d = 0; d < 6; ++d) in7[d] = xh[node * 9 + 3 + d] * m;
  in7[6] = ctx[node];
  float s = eb[c];
#pragma unroll
  for (int d = 0; d < 7; ++d) s += in7[d] * ew[d * HID + c];
  h[idx] = s;
  if (c < 3) {
    float xv = xh[node * 9 + c] * m;
    x0[node * 3 + c]  = xv;
    xb0[node * 3 + c] = xv;
  }
}

// ---------------------------------------------------------------------------
// GEMM core: 16 rows x 128 cols, K=128, prefetched B fragments
// ---------------------------------------------------------------------------
__device__ __forceinline__ void gemm_acc_step(v8f acc[8], v16h a,
                                              const v16h* __restrict__ W,
                                              int ks, int lane) {
  v16h bf[8];
#pragma unroll
  for (int nt = 0; nt < 8; ++nt) bf[nt] = W[(ks * 8 + nt) * 32 + lane];
#pragma unroll
  for (int nt = 0; nt < 8; ++nt)
    acc[nt] = __builtin_amdgcn_wmma_f32_16x16x32_f16(
        false, a, false, bf[nt], (short)0, acc[nt], false, false);
}

// out = f(A@W + bias) (+resid) (*nm)
__global__ __launch_bounds__(32) void gemm_node_kernel(
    const float* __restrict__ A, const v16h* __restrict__ W,
    const float* __restrict__ bias, const float* __restrict__ resid,
    const float* __restrict__ nmask, float* __restrict__ out, int act) {
  const int lane = threadIdx.x & 31;
  const int row0 = blockIdx.x * 16;

  v8f acc[8];
  v8f zero = {};
#pragma unroll
  for (int nt = 0; nt < 8; ++nt) acc[nt] = zero;

#pragma unroll
  for (int ks = 0; ks < 4; ++ks) {
    v16h a = frag_a_from_f32(A + (size_t)row0 * HID, HID, ks * 32, lane);
    gemm_acc_step(acc, a, W, ks, lane);
  }

  const int n = lane & 15, hf = lane >> 4;
#pragma unroll
  for (int nt = 0; nt < 8; ++nt) {
    int col = nt * 16 + n;
    float bv = bias ? bias[col] : 0.0f;
#pragma unroll
    for (int v = 0; v < 8; ++v) {
      int row = row0 + v + 8 * hf;
      float val = acc[nt][v] + bv;
      if (act) val = silu_f(val);
      if (resid) val += resid[(size_t)row * HID + col];
      if (nmask) val *= nmask[row];
      out[(size_t)row * HID + col] = val;
    }
  }
}

// out = silu(A@W1 + A2@W2 + bias)   (node-update stage 1, branch-free)
__global__ __launch_bounds__(32) void gemm_node_dual_kernel(
    const float* __restrict__ A, const v16h* __restrict__ W1,
    const float* __restrict__ A2, const v16h* __restrict__ W2,
    const float* __restrict__ bias, float* __restrict__ out) {
  const int lane = threadIdx.x & 31;
  const int row0 = blockIdx.x * 16;

  v8f acc[8];
  v8f zero = {};
#pragma unroll
  for (int nt = 0; nt < 8; ++nt) acc[nt] = zero;

#pragma unroll
  for (int ks = 0; ks < 4; ++ks) {
    v16h a1 = frag_a_from_f32(A  + (size_t)row0 * HID, HID, ks * 32, lane);
    v16h a2 = frag_a_from_f32(A2 + (size_t)row0 * HID, HID, ks * 32, lane);
    gemm_acc_step(acc, a1, W1, ks, lane);
    gemm_acc_step(acc, a2, W2, ks, lane);
  }

  const int n = lane & 15, hf = lane >> 4;
#pragma unroll
  for (int nt = 0; nt < 8; ++nt) {
    int col = nt * 16 + n;
    float bv = bias[col];
#pragma unroll
    for (int v = 0; v < 8; ++v) {
      int row = row0 + v + 8 * hf;
      out[(size_t)row * HID + col] = silu_f(acc[nt][v] + bv);
    }
  }
}

// out1 = A@W1 ; out2 = A@W2    (shared A fragments: hA/hB projections)
__global__ __launch_bounds__(32) void gemm_dual_out_kernel(
    const float* __restrict__ A, const v16h* __restrict__ W1,
    const v16h* __restrict__ W2, float* __restrict__ out1,
    float* __restrict__ out2) {
  const int lane = threadIdx.x & 31;
  const int row0 = blockIdx.x * 16;

  v8f acc1[8], acc2[8];
  v8f zero = {};
#pragma unroll
  for (int nt = 0; nt < 8; ++nt) { acc1[nt] = zero; acc2[nt] = zero; }

#pragma unroll
  for (int ks = 0; ks < 4; ++ks) {
    v16h a = frag_a_from_f32(A + (size_t)row0 * HID, HID, ks * 32, lane);
    gemm_acc_step(acc1, a, W1, ks, lane);
    gemm_acc_step(acc2, a, W2, ks, lane);
  }

  const int n = lane & 15, hf = lane >> 4;
#pragma unroll
  for (int nt = 0; nt < 8; ++nt) {
    int col = nt * 16 + n;
#pragma unroll
    for (int v = 0; v < 8; ++v) {
      int row = row0 + v + 8 * hf;
      out1[(size_t)row * HID + col] = acc1[nt][v];
      out2[(size_t)row * HID + col] = acc2[nt][v];
    }
  }
}

// ---------------------------------------------------------------------------
// Edge MLP, fused per (batch b, node i): for all j
//   pre  = hA[i] + hB[j] + radial(x)*w1r + dist0(x0)*w1d + b1
//   m    = silu(silu(pre) @ e_w2 + e_b2) * em
//   agg[i] = sum_j m / NORM
// 6 waves, each owns a 16-row j tile; deterministic column reduction.
// ---------------------------------------------------------------------------
__global__ __launch_bounds__(192) void edge_mlp_kernel(
    const float* __restrict__ hA, const float* __restrict__ hB,
    const float* __restrict__ xc, const float* __restrict__ x0,
    const float* __restrict__ w1r, const float* __restrict__ w1d,
    const float* __restrict__ b1,
    const v16h* __restrict__ W2, const float* __restrict__ b2,
    const float* __restrict__ emask, float* __restrict__ agg) {
  const int bi = blockIdx.x;
  const int b = bi / NN;
  const int in_ = bi - b * NN;
  const int tid = threadIdx.x;
  const int w = tid >> 5, lane = tid & 31;

  __shared__ __align__(16) _Float16 tS[6 * 16 * HID];  // silu(pre) tiles (24 KB)
  __shared__ float partS[6 * 2 * HID];                 // column partial sums

  float hAi[4], wr[4], wd[4], bb[4];
#pragma unroll
  for (int cc = 0; cc < 4; ++cc) {
    int c = lane + 32 * cc;
    hAi[cc] = hA[(size_t)bi * HID + c];
    wr[cc] = w1r[c];
    wd[cc] = w1d[c];
    bb[cc] = b1[c];
  }
  const float xi0 = xc[bi * 3 + 0], xi1 = xc[bi * 3 + 1], xi2 = xc[bi * 3 + 2];
  const float oi0 = x0[bi * 3 + 0], oi1 = x0[bi * 3 + 1], oi2 = x0[bi * 3 + 2];

  _Float16* tw = tS + w * 16 * HID;
  for (int jj = 0; jj < 16; ++jj) {
    int j = w * 16 + jj;
    int nj = b * NN + j;
    float dx0 = xi0 - xc[nj * 3 + 0];
    float dx1 = xi1 - xc[nj * 3 + 1];
    float dx2 = xi2 - xc[nj * 3 + 2];
    float rad = dx0 * dx0 + dx1 * dx1 + dx2 * dx2;
    float e0 = oi0 - x0[nj * 3 + 0];
    float e1 = oi1 - x0[nj * 3 + 1];
    float e2 = oi2 - x0[nj * 3 + 2];
    float d0 = e0 * e0 + e1 * e1 + e2 * e2;
#pragma unroll
    for (int cc = 0; cc < 4; ++cc) {
      int c = lane + 32 * cc;
      float pre = hAi[cc] + hB[(size_t)nj * HID + c] + rad * wr[cc] + d0 * wd[cc] + bb[cc];
      tw[jj * HID + c] = (_Float16)silu_f(pre);
    }
  }
  __syncthreads();

  v8f acc[8];
  v8f zero = {};
#pragma unroll
  for (int nt = 0; nt < 8; ++nt) acc[nt] = zero;
#pragma unroll
  for (int ks = 0; ks < 4; ++ks) {
    v16h a = frag_a_from_h16(tw, HID, ks * 32, lane);
    gemm_acc_step(acc, a, W2, ks, lane);
  }

  const int n = lane & 15, hf = lane >> 4;
  float em[8];
#pragma unroll
  for (int v = 0; v < 8; ++v) {
    int j = w * 16 + v + 8 * hf;
    em[v] = emask[(size_t)b * NEDGE + in_ * NN + j];
  }
#pragma unroll
  for (int nt = 0; nt < 8; ++nt) {
    int col = nt * 16 + n;
    float bv = b2[col];
    float cs = 0.0f;
#pragma unroll
    for (int v = 0; v < 8; ++v) cs += silu_f(acc[nt][v] + bv) * em[v];
    partS[(w * 2 + hf) * HID + col] = cs;
  }
  __syncthreads();
  if (tid < HID) {
    float s = 0.0f;
#pragma unroll
    for (int q = 0; q < 12; ++q) s += partS[q * HID + tid];
    agg[(size_t)bi * HID + tid] = s * (1.0f / NORMV);
  }
}

// ---------------------------------------------------------------------------
// Coordinate MLP, fused per (b, i):
//   q   = silu(silu(pre) @ c_w2 + c_b2)
//   phi = q @ c_w3 ; x' = x + sum_j coord_diff*phi*em / NORM
// ---------------------------------------------------------------------------
__global__ __launch_bounds__(192) void coord_mlp_kernel(
    const float* __restrict__ hA, const float* __restrict__ hB,
    const float* __restrict__ xc, const float* __restrict__ x0,
    const float* __restrict__ w1r, const float* __restrict__ w1d,
    const float* __restrict__ b1,
    const v16h* __restrict__ W2, const float* __restrict__ b2,
    const float* __restrict__ w3, const float* __restrict__ emask,
    float* __restrict__ xn) {
  const int bi = blockIdx.x;
  const int b = bi / NN;
  const int in_ = bi - b * NN;
  const int tid = threadIdx.x;
  const int w = tid >> 5, lane = tid & 31;

  __shared__ __align__(16) unsigned char smemRaw[6 * 16 * HID * sizeof(float)]; // 48 KB
  __shared__ float cS[NN * 3];   // per-j coordinate contributions
  _Float16* tS = (_Float16*)smemRaw;
  float* qS = (float*)smemRaw;

  float hAi[4], wr[4], wd[4], bb[4];
#pragma unroll
  for (int cc = 0; cc < 4; ++cc) {
    int c = lane + 32 * cc;
    hAi[cc] = hA[(size_t)bi * HID + c];
    wr[cc] = w1r[c];
    wd[cc] = w1d[c];
    bb[cc] = b1[c];
  }
  const float xi0 = xc[bi * 3 + 0], xi1 = xc[bi * 3 + 1], xi2 = xc[bi * 3 + 2];
  const float oi0 = x0[bi * 3 + 0], oi1 = x0[bi * 3 + 1], oi2 = x0[bi * 3 + 2];

  _Float16* tw = tS + w * 16 * HID;
  for (int jj = 0; jj < 16; ++jj) {
    int j = w * 16 + jj;
    int nj = b * NN + j;
    float dx0 = xi0 - xc[nj * 3 + 0];
    float dx1 = xi1 - xc[nj * 3 + 1];
    float dx2 = xi2 - xc[nj * 3 + 2];
    float rad = dx0 * dx0 + dx1 * dx1 + dx2 * dx2;
    float e0 = oi0 - x0[nj * 3 + 0];
    float e1 = oi1 - x0[nj * 3 + 1];
    float e2 = oi2 - x0[nj * 3 + 2];
    float d0 = e0 * e0 + e1 * e1 + e2 * e2;
#pragma unroll
    for (int cc = 0; cc < 4; ++cc) {
      int c = lane + 32 * cc;
      float pre = hAi[cc] + hB[(size_t)nj * HID + c] + rad * wr[cc] + d0 * wd[cc] + bb[cc];
      tw[jj * HID + c] = (_Float16)silu_f(pre);
    }
  }
  __syncthreads();

  v8f acc[8];
  v8f zero = {};
#pragma unroll
  for (int nt = 0; nt < 8; ++nt) acc[nt] = zero;
#pragma unroll
  for (int ks = 0; ks < 4; ++ks) {
    v16h a = frag_a_from_h16(tw, HID, ks * 32, lane);
    gemm_acc_step(acc, a, W2, ks, lane);
  }
  __syncthreads();  // all waves done reading tS before it is reused as qS

  const int n = lane & 15, hf = lane >> 4;
#pragma unroll
  for (int nt = 0; nt < 8; ++nt) {
    int col = nt * 16 + n;
    float bv = b2[col];
#pragma unroll
    for (int v = 0; v < 8; ++v) {
      int j = w * 16 + v + 8 * hf;
      qS[j * HID + col] = silu_f(acc[nt][v] + bv);
    }
  }
  __syncthreads();

  if (tid < NN) {
    int j = tid;
    int nj = b * NN + j;
    float phi = 0.0f;
    for (int c = 0; c < HID; ++c) phi += qS[j * HID + c] * w3[c];
    float em = emask[(size_t)b * NEDGE + in_ * NN + j];
    float dx0 = xi0 - xc[nj * 3 + 0];
    float dx1 = xi1 - xc[nj * 3 + 1];
    float dx2 = xi2 - xc[nj * 3 + 2];
    float rad = dx0 * dx0 + dx1 * dx1 + dx2 * dx2;
    float inv = rsqrtf(rad + 1e-8f);
    float pe = phi * em * inv;
    cS[j * 3 + 0] = dx0 * pe;
    cS[j * 3 + 1] = dx1 * pe;
    cS[j * 3 + 2] = dx2 * pe;
  }
  __syncthreads();
  if (tid < 3) {
    float s = 0.0f;
    for (int j = 0; j < NN; ++j) s += cS[j * 3 + tid];  // fixed order
    xn[bi * 3 + tid] = xc[bi * 3 + tid] + s * (1.0f / NORMV);
  }
}

// ---------------------------------------------------------------------------
// Output head (K=128 -> 5):  hf = (u @ f_w2 + f_b2) * nm
// ---------------------------------------------------------------------------
__global__ void head_kernel(const float* __restrict__ u,
                            const float* __restrict__ fw2,
                            const float* __restrict__ fb2,
                            const float* __restrict__ nmask,
                            float* __restrict__ hf) {
  int idx = blockIdx.x * blockDim.x + threadIdx.x;
  if (idx >= BNODES * 5) return;
  int node = idx / 5;
  int o = idx - node * 5;
  float s = fb2[o];
  for (int c = 0; c < HID; ++c) s += u[(size_t)node * HID + c] * fw2[c * 5 + o];
  hf[idx] = s * nmask[node];
}

// ---------------------------------------------------------------------------
// Final outputs: vel (mean-subtracted), vel_std, h_mean, h_std
// ---------------------------------------------------------------------------
__global__ __launch_bounds__(128) void finalize_kernel(
    const float* __restrict__ x, const float* __restrict__ nmask,
    const float* __restrict__ hf, float* __restrict__ out) {
  const int b = blockIdx.x;
  const int tid = threadIdx.x;
  __shared__ float stage[NN * 5];
  __shared__ float sums[5];  // nn, velsum[3], s0
  if (tid < NN) {
    int node = b * NN + tid;
    float m = nmask[node];
    stage[tid * 5 + 0] = m;
    stage[tid * 5 + 1] = x[node * 3 + 0] * m;
    stage[tid * 5 + 2] = x[node * 3 + 1] * m;
    stage[tid * 5 + 3] = x[node * 3 + 2] * m;
    stage[tid * 5 + 4] = hf[node * 5 + 0];
  }
  __syncthreads();
  if (tid < 5) {
    float s = 0.0f;
    for (int j = 0; j < NN; ++j) s += stage[j * 5 + tid];  // fixed order
    sums[tid] = s;
  }
  __syncthreads();
  if (tid < NN) {
    int node = b * NN + tid;
    float m = nmask[node];
    float nn = sums[0];
    float* vel = out;                     // (B,N,3)
    float* vstd = out + BNODES * 3;       // (B,N,1)
    float* hmean = vstd + BNODES;         // (B,N,2)
    float* hstd = hmean + BNODES * 2;     // (B,N,2)
#pragma unroll
    for (int d = 0; d < 3; ++d)
      vel[node * 3 + d] = x[node * 3 + d] * m - (sums[1 + d] / nn) * m;
    vstd[node] = expf(0.5f * sums[4]);
    hmean[node * 2 + 0] = hf[node * 5 + 1];
    hmean[node * 2 + 1] = hf[node * 5 + 2];
    hstd[node * 2 + 0] = expf(0.5f * hf[node * 5 + 3]);
    hstd[node * 2 + 1] = expf(0.5f * hf[node * 5 + 4]);
  }
}

// ===========================================================================
// Host orchestration
// ===========================================================================
extern "C" void kernel_launch(void* const* d_in, const int* in_sizes, int n_in,
                              void* d_out, int out_size, void* d_ws, size_t ws_size,
                              hipStream_t stream) {
  (void)in_sizes; (void)n_in; (void)out_size; (void)ws_size;

  const float* xh    = (const float*)d_in[0];
  // d_in[1] = bonds_edge_attr (unused, all zeros)
  const float* nmv   = (const float*)d_in[2];
  const float* emv   = (const float*)d_in[3];
  const float* ctx   = (const float*)d_in[4];
  const float* emb_w = (const float*)d_in[5];
  const float* emb_b = (const float*)d_in[6];
  const float* e_w1  = (const float*)d_in[7];
  const float* e_b1  = (const float*)d_in[8];
  const float* e_w2  = (const float*)d_in[9];
  const float* e_b2  = (const float*)d_in[10];
  const float* n_w1  = (const float*)d_in[11];
  const float* n_b1  = (const float*)d_in[12];
  const float* n_w2  = (const float*)d_in[13];
  const float* n_b2  = (const float*)d_in[14];
  const float* c_w1  = (const float*)d_in[15];
  const float* c_b1  = (const float*)d_in[16];
  const float* c_w2  = (const float*)d_in[17];
  const float* c_b2  = (const float*)d_in[18];
  const float* c_w3  = (const float*)d_in[19];
  const float* out_w = (const float*)d_in[20];
  const float* out_b = (const float*)d_in[21];
  const float* f_w1  = (const float*)d_in[22];
  const float* f_b1  = (const float*)d_in[23];
  const float* f_w2  = (const float*)d_in[24];
  const float* f_b2  = (const float*)d_in[25];

  // ---- workspace carve-up (~7 MB) ----
  char* ws = (char*)d_ws;
  size_t off = 0;
  auto carve = [&](size_t bytes) -> void* {
    void* p = (void*)(ws + off);
    off += (bytes + 255) & ~(size_t)255;
    return p;
  };
  float* x0f  = (float*)carve(BNODES * 3 * sizeof(float));
  float* xb0v = (float*)carve(BNODES * 3 * sizeof(float));
  float* xb1v = (float*)carve(BNODES * 3 * sizeof(float));
  float* h0v  = (float*)carve((size_t)BNODES * HID * sizeof(float));
  float* h1v  = (float*)carve((size_t)BNODES * HID * sizeof(float));
  float* hAb  = (float*)carve((size_t)BNODES * HID * sizeof(float));
  float* hBb  = (float*)carve((size_t)BNODES * HID * sizeof(float));
  float* tmpb = (float*)carve((size_t)BNODES * HID * sizeof(float));
  float* aggb = (float*)carve((size_t)BNODES * HID * sizeof(float));
  float* hfb  = (float*)carve(BNODES * 5 * sizeof(float));
  const size_t MATB = 1024 * sizeof(v16h);  // packed 128x128 matrix = 32 KB
  v16h* ew1A_p = (v16h*)carve(8 * MATB);
  v16h* ew1B_p = (v16h*)carve(8 * MATB);
  v16h* ew2_p  = (v16h*)carve(8 * MATB);
  v16h* nw1A_p = (v16h*)carve(8 * MATB);
  v16h* nw1B_p = (v16h*)carve(8 * MATB);
  v16h* nw2_p  = (v16h*)carve(8 * MATB);
  v16h* cw1A_p = (v16h*)carve(4 * MATB);
  v16h* cw1B_p = (v16h*)carve(4 * MATB);
  v16h* cw2_p  = (v16h*)carve(4 * MATB);
  v16h* outw_p = (v16h*)carve(1 * MATB);
  v16h* fw1_p  = (v16h*)carve(1 * MATB);

  // ---- pack weights into B-fragment layout ----
  auto pack = [&](const float* src, size_t stride_f, v16h* dst, int nmat) {
    int total = nmat * 1024;
    pack_w_kernel<<<(total + 255) / 256, 256, 0, stream>>>(src, stride_f, dst, nmat);
  };
  pack(e_w1,             (size_t)258 * HID, ew1A_p, 8);
  pack(e_w1 + 128 * HID, (size_t)258 * HID, ew1B_p, 8);
  pack(e_w2,             (size_t)HID * HID, ew2_p, 8);
  pack(n_w1,             (size_t)256 * HID, nw1A_p, 8);
  pack(n_w1 + 128 * HID, (size_t)256 * HID, nw1B_p, 8);
  pack(n_w2,             (size_t)HID * HID, nw2_p, 8);
  pack(c_w1,             (size_t)258 * HID, cw1A_p, 4);
  pack(c_w1 + 128 * HID, (size_t)258 * HID, cw1B_p, 4);
  pack(c_w2,             (size_t)HID * HID, cw2_p, 4);
  pack(out_w, 0, outw_p, 1);
  pack(f_w1,  0, fw1_p, 1);

  // ---- embedding + coordinate init ----
  embed_init_kernel<<<(BNODES * HID + 255) / 256, 256, 0, stream>>>(
      xh, nmv, ctx, emb_w, emb_b, h0v, x0f, xb0v);

  const int MT = BNODES / 16;  // 96 M-tiles
  float* hcur = h0v;
  float* hnext = h1v;

  for (int l = 0; l < 4; ++l) {
    float* xc = (l & 1) ? xb1v : xb0v;
    float* xn = (l & 1) ? xb0v : xb1v;

    for (int s = 0; s < 2; ++s) {
      int i = l * 2 + s;
      // hA = h @ e_w1[:H],  hB = h @ e_w1[H:2H]  (shared A fragments)
      gemm_dual_out_kernel<<<MT, 32, 0, stream>>>(
          hcur, ew1A_p + (size_t)i * 1024, ew1B_p + (size_t)i * 1024, hAb, hBb);
      // fused edge MLP -> agg
      edge_mlp_kernel<<<BNODES, 192, 0, stream>>>(
          hAb, hBb, xc, x0f,
          e_w1 + (size_t)i * 258 * HID + 256 * HID,
          e_w1 + (size_t)i * 258 * HID + 257 * HID,
          e_b1 + i * HID,
          ew2_p + (size_t)i * 1024, e_b2 + i * HID, emv, aggb);
      // node update: tmp = silu(h@Wa + agg@Wb + b1) ; h = (h + tmp@W2 + b2)*nm
      gemm_node_dual_kernel<<<MT, 32, 0, stream>>>(
          hcur, nw1A_p + (size_t)i * 1024, aggb, nw1B_p + (size_t)i * 1024,
          n_b1 + i * HID, tmpb);
      gemm_node_kernel<<<MT, 32, 0, stream>>>(
          tmpb, nw2_p + (size_t)i * 1024, n_b2 + i * HID, hcur, nmv, hnext, 0);
      float* t = hcur; hcur = hnext; hnext = t;
    }

    // coordinate layer l
    gemm_dual_out_kernel<<<MT, 32, 0, stream>>>(
        hcur, cw1A_p + (size_t)l * 1024, cw1B_p + (size_t)l * 1024, hAb, hBb);
    coord_mlp_kernel<<<BNODES, 192, 0, stream>>>(
        hAb, hBb, xc, x0f,
        c_w1 + (size_t)l * 258 * HID + 256 * HID,
        c_w1 + (size_t)l * 258 * HID + 257 * HID,
        c_b1 + l * HID,
        cw2_p + (size_t)l * 1024, c_b2 + l * HID,
        c_w3 + l * HID, emv, xn);
  }
  // after 4 layers the coordinates land back in xb0v

  // ---- output head ----
  gemm_node_kernel<<<MT, 32, 0, stream>>>(hcur, outw_p, out_b, nullptr, nmv,
                                          hAb, 0);                          // h_out
  gemm_node_kernel<<<MT, 32, 0, stream>>>(hAb, fw1_p, f_b1, nullptr, nullptr,
                                          tmpb, 1);                         // silu
  head_kernel<<<(BNODES * 5 + 255) / 256, 256, 0, stream>>>(tmpb, f_w2, f_b2, nmv, hfb);
  finalize_kernel<<<NB, 128, 0, stream>>>(xb0v, nmv, hfb, (float*)d_out);
}
